// TPCL_62122406969664
// MI455X (gfx1250) — matrix operand order repaired and checked by reading, not compiled, not run on hardware
//
#include <hip/hip_runtime.h>

typedef __attribute__((ext_vector_type(16))) _Float16 v16h;
typedef __attribute__((ext_vector_type(8)))  _Float16 v8h;
typedef __attribute__((ext_vector_type(8)))  float    v8f;
typedef __attribute__((__vector_size__(4 * sizeof(int)))) int v4i;

#define NODE_F 80   // 32 + 48
#define FE     32
#define HDIM   128
#define WN     2304
#define HPAD   128  // f16 row stride for h tile
#define WAVES  8    // waves (16-edge tiles) per block
#define NCHUNK 144  // 2304 / 16
#define GRP    8    // chunks per staged B panel
#define NGRP   (NCHUNK / GRP)            // 18 groups
#define PANEL_HALVES (GRP * 16 * HDIM)   // 16384 halves = 32 KB per buffer

__device__ __forceinline__ v8f wmma16(v16h a, v16h b, v8f c) {
  // D = A(16x32 f16) * B(32x16 f16) + C(16x16 f32)
  return __builtin_amdgcn_wmma_f32_16x16x32_f16(false, a, false, b, (short)0, c,
                                                false, false);
}

__device__ __forceinline__ void lds_fence() {
  asm volatile("s_wait_dscnt 0x0" ::: "memory");
}

// Wait until at most `n` async-LDS ops of this wave remain outstanding.
template <int NREM>
__device__ __forceinline__ void wait_async() {
#if defined(__AMDGCN__) && __has_builtin(__builtin_amdgcn_s_wait_asynccnt)
  __builtin_amdgcn_s_wait_asynccnt(NREM);
#elif defined(__AMDGCN__)
  asm volatile("s_wait_asynccnt %0" ::"i"(NREM) : "memory");
#endif
}

// 16-byte global -> LDS copy via the gfx1250 async-DMA path.
__device__ __forceinline__ void copy16_to_lds(const _Float16* g, _Float16* l) {
#if defined(__AMDGCN__) && __has_builtin(__builtin_amdgcn_global_load_async_to_lds_b128)
  typedef __attribute__((address_space(1))) v4i* gp_t;
  typedef __attribute__((address_space(3))) v4i* lp_t;
  __builtin_amdgcn_global_load_async_to_lds_b128((gp_t)g, (lp_t)l, 0, 0);
#else
  *(v8h*)l = *(const v8h*)g;
#endif
}

struct alignas(16) WaveLds {
  float    x0a[16 * 32];  // A0*sh0[e]*x0[e,u]            (w1 path)
  float    x0b[16 * 32];  // (A1/sqrt3)*x0[e,u]           (w2 path)
  float    y1a[16 * 16];  // (A0/sqrt3)*sum_m x1*sh1      (w4 path)
  float    x1b[16 * 48];  // (A1/sqrt3)*sh0*x1[e,u,m3]    (w3 path)
  _Float16 h[16 * HPAD];  // hidden activations, f16
};

__global__ void zero_kernel(float* out, float* cnt, int n_out, int n_cnt) {
  int i = blockIdx.x * blockDim.x + threadIdx.x;
  if (i < n_out) out[i] = 0.f;
  if (i < n_cnt) cnt[i] = 0.f;
}

// Transpose + f16-convert MLP weights: w1t[c][k] (128x32), w2t[c][k] (2304x128)
__global__ void prep_kernel(const float* __restrict__ w1, const float* __restrict__ w2,
                            _Float16* __restrict__ w1t, _Float16* __restrict__ w2t) {
  int i = blockIdx.x * blockDim.x + threadIdx.x;
  if (i < FE * HDIM) {
    int c = i >> 5, k = i & 31;
    w1t[i] = (_Float16)w1[k * HDIM + c];
  }
  int j = i - FE * HDIM;
  if (j >= 0 && j < WN * HDIM) {
    int c = j >> 7, k = j & 127;
    w2t[j] = (_Float16)w2[k * WN + c];
  }
}

__global__ void finalize_kernel(float* __restrict__ out, const float* __restrict__ cnt,
                                int n) {
  int i = blockIdx.x * blockDim.x + threadIdx.x;
  if (i < n) out[i] = out[i] / fmaxf(cnt[i / NODE_F], 1.0f);
}

__global__ __launch_bounds__(WAVES * 32)
void edge_kernel(const float* __restrict__ node_attr,
                 const int*   __restrict__ edge_index,
                 const float* __restrict__ edge_attr,
                 const float* __restrict__ edge_sh,
                 const float* __restrict__ fc_b1,
                 const float* __restrict__ fc_b2,
                 const _Float16* __restrict__ w1t,
                 const _Float16* __restrict__ w2t,
                 float* __restrict__ out,
                 float* __restrict__ cnt,
                 int E) {
  extern __shared__ char dynsmem[];
  WaveLds*  wls    = (WaveLds*)dynsmem;
  _Float16* panelA = (_Float16*)(dynsmem + WAVES * sizeof(WaveLds));
  _Float16* panelB = panelA + PANEL_HALVES;

  const int lane = threadIdx.x & 31;
  const int wave = threadIdx.x >> 5;
  const int tile = blockIdx.x * WAVES + wave;
  const int e0   = tile * 16;          // may be >= E: wave still joins barriers
  WaveLds& L = wls[wave];
  const int n    = lane & 15;
  const int hi   = lane >> 4;
  const int rowb = 8 * hi;

  // Cooperative async stage of one 32 KB contiguous B panel (8 chunks).
  auto stage = [&](int g, _Float16* dst) {
    const _Float16* gbase = w2t + (size_t)g * GRP * 16 * HDIM;
    for (int i = threadIdx.x; i < PANEL_HALVES / 8; i += WAVES * 32)
      copy16_to_lds(gbase + i * 8, dst + i * 8);
  };

  // Kick off group 0 DMA immediately; it overlaps all the prologue work below.
  stage(0, panelA);

  constexpr float A0c = 0.14433756729740643f;  // sqrt(1/48)
  constexpr float A0I = 0.08333333333333333f;  // sqrt(1/48)/sqrt(3) = 1/12
  constexpr float A1I = 0.14433756729740643f;  // sqrt(3/48)/sqrt(3)

  // ---------------- stage per-edge TP coefficients into LDS ----------------
  for (int i = lane; i < 16 * 32; i += 32) {
    int m = i >> 5, u = i & 31;
    int e = e0 + m;
    float x0 = 0.f, s0 = 0.f;
    if (e < E) {
      int d = edge_index[E + e];         // edge_dst
      s0 = edge_sh[e * 4 + 0];
      x0 = node_attr[d * NODE_F + u];
    }
    L.x0a[i] = A0c * s0 * x0;
    L.x0b[i] = A1I * x0;
  }
  for (int i = lane; i < 16 * 16; i += 32) {
    int m = i >> 4, u = i & 15;
    int e = e0 + m;
    float y = 0.f;
    if (e < E) {
      int d = edge_index[E + e];
      const float* xr = node_attr + d * NODE_F + 32 + u * 3;
      y = xr[0] * edge_sh[e * 4 + 1] + xr[1] * edge_sh[e * 4 + 2] +
          xr[2] * edge_sh[e * 4 + 3];
    }
    L.y1a[i] = A0I * y;
  }
  for (int i = lane; i < 16 * 48; i += 32) {
    int m = i / 48, j = i - m * 48;
    int e = e0 + m;
    float v = 0.f;
    if (e < E) {
      int d = edge_index[E + e];
      v = A1I * edge_sh[e * 4 + 0] * node_attr[d * NODE_F + 32 + j];
    }
    L.x1b[i] = v;
  }

  // sh1 per row (w2 path), register-resident
  float shr[8][3];
#pragma unroll
  for (int r = 0; r < 8; ++r) {
    int e = e0 + r + rowb;
#pragma unroll
    for (int m3 = 0; m3 < 3; ++m3)
      shr[r][m3] = (e < E) ? edge_sh[e * 4 + 1 + m3] : 0.f;
  }

  // ---------------- layer 1: h = relu(edge_attr @ W1 + b1) ----------------
  v16h af;
  {
    int e = e0 + n;
#pragma unroll
    for (int t = 0; t < 8; ++t) {
      float a = (e < E) ? edge_attr[e * FE + 16 * hi + t] : 0.f;
      float b = (e < E) ? edge_attr[e * FE + 16 * hi + 8 + t] : 0.f;
      af[t]     = (_Float16)a;
      af[8 + t] = (_Float16)b;
    }
  }
#pragma unroll 1
  for (int cc = 0; cc < 8; ++cc) {
    const _Float16* bp = w1t + (cc * 16 + n) * FE + 16 * hi;
    v16h bf;
#pragma unroll
    for (int t = 0; t < 16; ++t) bf[t] = bp[t];
    v8f d = {};
    d = wmma16(af, bf, d);
    float b1v = fc_b1[cc * 16 + n];
#pragma unroll
    for (int r = 0; r < 8; ++r) {
      float hv = fmaxf(d[r] + b1v, 0.f);
      L.h[(r + rowb) * HPAD + cc * 16 + n] = (_Float16)hv;  // D-layout scatter
    }
  }

  lds_fence();  // intra-wave: LDS stores (coefs + h) visible to all lanes

  // h as 4 A fragments (K = 0..127)
  v16h ha[4];
#pragma unroll
  for (int kb = 0; kb < 4; ++kb) {
    const _Float16* hp = &L.h[n * HPAD + kb * 32 + 8 * hi];
    v8h lo  = *(const v8h*)hp;
    v8h hi8 = *(const v8h*)(hp + 16);
#pragma unroll
    for (int t = 0; t < 8; ++t) { ha[kb][t] = lo[t]; ha[kb][8 + t] = hi8[t]; }
  }

  // ---------------- layer 2 fused with tensor product ----------------
  float acc0[2][8];   // out0 cols n, 16+n   (per row r)
  float acc1[8][3];   // out1 (v=n, m3)      (per row r)
#pragma unroll
  for (int r = 0; r < 8; ++r) {
    acc0[0][r] = 0.f; acc0[1][r] = 0.f;
    acc1[r][0] = 0.f; acc1[r][1] = 0.f; acc1[r][2] = 0.f;
  }

  // 16x16 chunk of w = h @ W2 + b2; B fragments from the given LDS panel.
  auto w_chunk = [&](const _Float16* pan, int lp, int c0) -> v8f {
    float bb = fc_b2[(c0 + lp) * 16 + n];
    v8f d;
#pragma unroll
    for (int r = 0; r < 8; ++r) d[r] = bb;
#pragma unroll
    for (int kb = 0; kb < 4; ++kb) {
      const _Float16* bp = pan + (lp * 16 + n) * HDIM + kb * 32 + 8 * hi;
      v8h b0 = *(const v8h*)bp;
      v8h b1 = *(const v8h*)(bp + 16);
      v16h bf;
#pragma unroll
      for (int t = 0; t < 8; ++t) { bf[t] = b0[t]; bf[8 + t] = b1[t]; }
      d = wmma16(ha[kb], bf, d);
    }
    return d;
  };

  // Software-pipelined over panel groups: DMA of group g+1 overlaps WMMAs of g.
#pragma unroll 1
  for (int g = 0; g < NGRP; ++g) {
    const int c0 = g * GRP;
    const _Float16* cur = (g & 1) ? panelB : panelA;
    _Float16*       nxt = (g & 1) ? panelA : panelB;

    if (g + 1 < NGRP) {
      stage(g + 1, nxt);       // 8 async b128s per thread, in flight during compute
      wait_async<8>();         // in-order completion: oldest 8 (= group g) done
    } else {
      wait_async<0>();         // drain last group
    }
    __syncthreads();           // group g panel visible to all waves

    if (c0 < 64) {
      // w1: cols [0,1024). chunk pair -> u fixed, v = n / 16+n
#pragma unroll 1
      for (int p = 0; p < 4; ++p) {
        v8f d0 = w_chunk(cur, 2 * p, c0);
        v8f d1 = w_chunk(cur, 2 * p + 1, c0);
        int u = (c0 >> 1) + p;
#pragma unroll
        for (int r = 0; r < 8; ++r) {
          float cf = L.x0a[(r + rowb) * 32 + u];
          acc0[0][r] += cf * d0[r];
          acc0[1][r] += cf * d1[r];
        }
      }
    } else if (c0 < 96) {
      // w2: cols [1024,1536). out1[v=n][m3] += x0b*sh1[m3]*w
#pragma unroll 1
      for (int p = 0; p < GRP; ++p) {
        v8f d = w_chunk(cur, p, c0);
        int u = c0 - 64 + p;
#pragma unroll
        for (int r = 0; r < 8; ++r) {
          float v = L.x0b[(r + rowb) * 32 + u] * d[r];
          acc1[r][0] += v * shr[r][0];
          acc1[r][1] += v * shr[r][1];
          acc1[r][2] += v * shr[r][2];
        }
      }
    } else if (c0 < 112) {
      // w3: cols [1536,1792). out1[v=n][m3] += x1b[u][m3]*w
#pragma unroll 1
      for (int p = 0; p < GRP; ++p) {
        v8f d = w_chunk(cur, p, c0);
        int u = c0 - 96 + p;
#pragma unroll
        for (int r = 0; r < 8; ++r) {
          const float* px = &L.x1b[((r + rowb) * 16 + u) * 3];
          acc1[r][0] += px[0] * d[r];
          acc1[r][1] += px[1] * d[r];
          acc1[r][2] += px[2] * d[r];
        }
      }
    } else {
      // w4: cols [1792,2304). chunk pair -> out0 += y1a*w
#pragma unroll 1
      for (int p = 0; p < 4; ++p) {
        v8f d0 = w_chunk(cur, 2 * p, c0);
        v8f d1 = w_chunk(cur, 2 * p + 1, c0);
        int u = ((c0 - 112) >> 1) + p;
#pragma unroll
        for (int r = 0; r < 8; ++r) {
          float cf = L.y1a[(r + rowb) * 16 + u];
          acc0[0][r] += cf * d0[r];
          acc0[1][r] += cf * d1[r];
        }
      }
    }
    __syncthreads();           // all waves done with `cur` before it is re-staged
  }

  // ---------------- scatter to nodes (segment sum via f32 atomics) ----------------
#pragma unroll
  for (int r = 0; r < 8; ++r) {
    int e = e0 + r + rowb;
    if (e < E) {
      int s = edge_index[e];             // edge_src
      float* op = out + (size_t)s * NODE_F;
      atomicAdd(op + n,      acc0[0][r]);
      atomicAdd(op + 16 + n, acc0[1][r]);
      float* o1 = op + 32 + 3 * n;
      atomicAdd(o1 + 0, acc1[r][0]);
      atomicAdd(o1 + 1, acc1[r][1]);
      atomicAdd(o1 + 2, acc1[r][2]);
    }
  }
  if (hi == 0) {
    int e = e0 + n;
    if (e < E) atomicAdd(cnt + edge_index[e], 1.0f);
  }
}

extern "C" void kernel_launch(void* const* d_in, const int* in_sizes, int n_in,
                              void* d_out, int out_size, void* d_ws, size_t ws_size,
                              hipStream_t stream) {
  const float* node_attr  = (const float*)d_in[0];
  const int*   edge_index = (const int*)d_in[1];
  const float* edge_attr  = (const float*)d_in[2];
  const float* edge_sh    = (const float*)d_in[3];
  const float* fc_w1      = (const float*)d_in[4];
  const float* fc_b1      = (const float*)d_in[5];
  const float* fc_w2      = (const float*)d_in[6];
  const float* fc_b2      = (const float*)d_in[7];
  float* out = (float*)d_out;

  const int N = in_sizes[0] / NODE_F;
  const int E = in_sizes[1] / 2;

  char* ws = (char*)d_ws;
  float* cnt = (float*)ws;
  size_t off = ((size_t)N * 4 + 255) / 256 * 256;
  _Float16* w1t = (_Float16*)(ws + off);
  _Float16* w2t = w1t + FE * HDIM;

  int n_out = N * NODE_F;
  zero_kernel<<<(n_out + 255) / 256, 256, 0, stream>>>(out, cnt, n_out, N);

  int n_prep = FE * HDIM + WN * HDIM;
  prep_kernel<<<(n_prep + 255) / 256, 256, 0, stream>>>(fc_w1, fc_w2, w1t, w2t);

  int tiles  = (E + 15) / 16;
  int blocks = (tiles + WAVES - 1) / WAVES;
  size_t shmem = WAVES * sizeof(WaveLds) + 2 * PANEL_HALVES * sizeof(_Float16);
  edge_kernel<<<blocks, WAVES * 32, shmem, stream>>>(node_attr, edge_index,
                                                     edge_attr, edge_sh, fc_b1,
                                                     fc_b2, w1t, w2t, out, cnt, E);

  finalize_kernel<<<(n_out + 255) / 256, 256, 0, stream>>>(out, cnt, n_out);
}
